// GPT_4629974745410
// MI455X (gfx1250) — matrix operand order repaired and checked by reading, not compiled, not run on hardware
//
#include <hip/hip_runtime.h>
#include <cstdint>

// ---------------- problem constants (GPT-2 small slice) ----------------
#define V_     50257
#define Tt     2048
#define Bb     2
#define Ee     768
#define Hh     12
#define HDm    64
#define FF_    3072
#define Ll     2
#define M_ROWS 4096          // B*T
#define GELU_FLAG 1
#define NT_FLAG   2

// ---------------- TDM availability probes ----------------
#if __has_builtin(__builtin_amdgcn_tensor_load_to_lds)
#  define HAVE_TDM 1
#else
#  define HAVE_TDM 0
#endif

// ---------------- WMMA types & helpers (gfx1250, wave32) ----------------
typedef __attribute__((ext_vector_type(16))) __bf16 v16bf;
typedef __attribute__((ext_vector_type(8)))  float  v8f;
typedef __attribute__((ext_vector_type(4)))  unsigned int ui32x4;
typedef __attribute__((ext_vector_type(8)))  int          i32x8;
typedef __attribute__((ext_vector_type(4)))  int          i32x4;

union BF16Frag { v16bf v; uint4 u[2]; };
union BF16x8   { uint4 u; __bf16 h[8]; };

__device__ __forceinline__ v8f wmma_bf16(v16bf a, v16bf b, v8f c) {
  // D = A(16x32 bf16) * B(32x16 bf16) + C(16x16 f32)
  return __builtin_amdgcn_wmma_f32_16x16x32_bf16(
      /*neg_a=*/false, a, /*neg_b=*/false, b,
      /*c_mod=*/(short)0, c, /*reuse_a=*/false, /*reuse_b=*/false);
}

// A-fragment: per-lane 16 contiguous bf16, K base = (lane/16)*16
__device__ __forceinline__ v16bf frag_a(const __bf16* p) {
  BF16Frag t;
  t.u[0] = *(const uint4*)(p);
  t.u[1] = *(const uint4*)(p + 8);
  return t.v;
}
// B-fragment: per-lane 16 contiguous bf16, K base = (lane/16)*16
__device__ __forceinline__ v16bf frag_b(const __bf16* p) {
  BF16Frag t;
  t.u[0] = *(const uint4*)(p);
  t.u[1] = *(const uint4*)(p + 8);
  return t.v;
}

__device__ __forceinline__ void wait_tensorcnt0() {
#if __has_builtin(__builtin_amdgcn_s_wait_tensorcnt)
  __builtin_amdgcn_s_wait_tensorcnt(0);
#else
  asm volatile("s_wait_tensorcnt 0x0" ::: "memory");
#endif
}

#if HAVE_TDM
typedef __attribute__((address_space(3))) const void lds_cv;
__device__ __forceinline__ uint32_t lds_offset(const void* p) {
  return (uint32_t)(unsigned long long)(lds_cv*)p;   // generic -> LDS offset
}

// TDM: DMA a [128 rows x 32 elem] bf16 tile (row pitch K elems) into LDS with a
// 16B pad after every 64B row -> LDS row stride 40 bf16 (matches fragment loads).
// Rows >= rowsAvail are hardware OOB -> written as zero by dim-1 bounds check.
__device__ __forceinline__ void tdm_load_tile(const __bf16* gsrc, uint32_t ldsOff,
                                              int rowsAvail, int K) {
  uint64_t ga = (uint64_t)(uintptr_t)gsrc;
  uint32_t td1 = rowsAvail < 0 ? 0u : (uint32_t)rowsAvail;
  ui32x4 g0;
  g0[0] = 1u;                                             // count=1 (valid D#)
  g0[1] = ldsOff;                                         // lds_addr
  g0[2] = (uint32_t)ga;                                   // global_addr[31:0]
  g0[3] = (uint32_t)((ga >> 32) & 0x01FFFFFFu) | (2u << 30); // addr[56:32] | type=2
  i32x8 g1;
  // data_size=1(2B) | pad_enable | pad_interval=3(16 DW=64B) | pad_amount=3(4 DW=16B)
  g1[0] = (int)((1u << 16) | (1u << 20) | (3u << 22) | (3u << 25));
  g1[1] = (int)(((uint32_t)K & 0xFFFFu) << 16);           // tensor_dim0[15:0]
  g1[2] = (int)((((uint32_t)K >> 16) & 0xFFFFu) | ((td1 & 0xFFFFu) << 16)); // td0 hi | td1 lo
  g1[3] = (int)(((td1 >> 16) & 0xFFFFu) | (32u << 16));   // td1 hi | tile_dim0=32
  g1[4] = (int)128u;                                      // tile_dim1=128, tile_dim2=0
  g1[5] = (int)(uint32_t)K;                               // tensor_dim0_stride lo32
  g1[6] = (int)(((uint32_t)((uint64_t)K >> 32)) & 0xFFFFu); // stride hi16 | td1_stride lo16=0
  g1[7] = 0;
  i32x4 z4 = {0, 0, 0, 0};
#if __has_include(<hip/amd_detail/amd_gfx1250_TDM.h>)
  i32x8 z8 = {0, 0, 0, 0, 0, 0, 0, 0};
  __builtin_amdgcn_tensor_load_to_lds(g0, g1, z4, z4, z8, 0);   // therock 6-arg form
#else
  __builtin_amdgcn_tensor_load_to_lds(g0, g1, z4, z4, 0);        // ROCm 7.2 5-arg form
#endif
}
#endif  // HAVE_TDM

// ---------------- small utility kernels ----------------
__global__ void cvt_bf16_kernel(const float* __restrict__ in, __bf16* __restrict__ out, int n) {
  int i = blockIdx.x * 256 + threadIdx.x;
  if (i < n) out[i] = (__bf16)in[i];
}

// in: [K][N] fp32 row-major  ->  out: [N][K] bf16 row-major (K-contiguous)
__global__ void transpose_bf16_kernel(const float* __restrict__ in, __bf16* __restrict__ out,
                                      int K, int N) {
  int i = blockIdx.x * 256 + threadIdx.x;
  if (i < K * N) {
    int k = i / N, n = i % N;
    out[(size_t)n * K + k] = (__bf16)in[i];
  }
}

__global__ void embed_kernel(const int* __restrict__ idx, const float* __restrict__ wte,
                             const float* __restrict__ wpe, float* __restrict__ x) {
  int row = blockIdx.x;            // b*T + t
  int t   = row & (Tt - 1);
  int tok = idx[row];
  const float* wt = wte + (size_t)tok * Ee;
  const float* wp = wpe + (size_t)t   * Ee;
  float* xr = x + (size_t)row * Ee;
  for (int i = threadIdx.x; i < Ee; i += 256) xr[i] = wt[i] + wp[i];
}

// per-row LayerNorm (E=768), fp32 in -> bf16 out
__global__ void layernorm_bf16_kernel(const float* __restrict__ x, const float* __restrict__ g,
                                      const float* __restrict__ b, __bf16* __restrict__ out) {
  int row = blockIdx.x, tid = threadIdx.x;
  __shared__ float red[256];
  const float* xr = x + (size_t)row * Ee;
  float v0 = xr[tid], v1 = xr[tid + 256], v2 = xr[tid + 512];
  red[tid] = v0 + v1 + v2;
  __syncthreads();
  for (int o = 128; o > 0; o >>= 1) { if (tid < o) red[tid] += red[tid + o]; __syncthreads(); }
  float mu = red[0] * (1.0f / Ee);
  __syncthreads();
  float d0 = v0 - mu, d1 = v1 - mu, d2 = v2 - mu;
  red[tid] = d0 * d0 + d1 * d1 + d2 * d2;
  __syncthreads();
  for (int o = 128; o > 0; o >>= 1) { if (tid < o) red[tid] += red[tid + o]; __syncthreads(); }
  float rstd = rsqrtf(red[0] * (1.0f / Ee) + 1e-5f);
  __bf16* orow = out + (size_t)row * Ee;
  orow[tid]       = (__bf16)(d0 * rstd * g[tid]       + b[tid]);
  orow[tid + 256] = (__bf16)(d1 * rstd * g[tid + 256] + b[tid + 256]);
  orow[tid + 512] = (__bf16)(d2 * rstd * g[tid + 512] + b[tid + 512]);
}

// ---------------- WMMA GEMM: C[M,N] = A[M,K](bf16) * Bt[N,K](bf16) ----------------
// 256 threads = 8 waves, block tile 128x128, K step 32; each wave: 64x32 (4x2 wmma tiles).
// Tiles staged by the Tensor Data Mover (double-buffered) when available.
__global__ __launch_bounds__(256) void gemm_bf16_kernel(
    const __bf16* __restrict__ A, const __bf16* __restrict__ Bt,
    const float* __restrict__ bias, const float* __restrict__ resid,
    float* __restrict__ Cf, __bf16* __restrict__ Cb,
    int M, int N, int K, int flags) {
#if HAVE_TDM
  __shared__ __align__(16) __bf16 As[2][128 * 40];
  __shared__ __align__(16) __bf16 Bs[2][128 * 40];
#else
  __shared__ __align__(16) __bf16 As1[128 * 40];
  __shared__ __align__(16) __bf16 Bs1[128 * 40];
#endif
  int tid  = threadIdx.x;
  int n0   = blockIdx.x * 128, m0 = blockIdx.y * 128;
  int lane = tid & 31, wid = tid >> 5;
  int wm   = wid >> 2, wn = wid & 3;              // 2x4 wave grid
  int half = lane >> 4, l16 = lane & 15;

  v8f acc[4][2];
  #pragma unroll
  for (int i = 0; i < 4; ++i)
    #pragma unroll
    for (int j = 0; j < 2; ++j)
      #pragma unroll
      for (int r = 0; r < 8; ++r) acc[i][j][r] = 0.0f;

  auto compute_step = [&](const __bf16* Asb, const __bf16* Bsb) {
    v16bf af[4], bfm[2];
    #pragma unroll
    for (int i = 0; i < 4; ++i)
      af[i] = frag_a(&Asb[(wm * 64 + i * 16 + l16) * 40 + half * 16]);
    #pragma unroll
    for (int j = 0; j < 2; ++j)
      bfm[j] = frag_b(&Bsb[(wn * 32 + j * 16 + l16) * 40 + half * 16]);
    #pragma unroll
    for (int i = 0; i < 4; ++i)
      #pragma unroll
      for (int j = 0; j < 2; ++j)
        acc[i][j] = wmma_bf16(af[i], bfm[j], acc[i][j]);
  };

#if HAVE_TDM
  // Uniform (scalar-branch) wave-0 predicate: TDM ignores EXEC, so keep it to 1 wave.
  bool issuer = ((__builtin_amdgcn_readfirstlane(tid) >> 5) == 0);
  int nk = K >> 5;
  int buf = 0;
  if (issuer) {
    tdm_load_tile(A  + (size_t)m0 * K, lds_offset(&As[0][0]), M - m0, K);
    tdm_load_tile(Bt + (size_t)n0 * K, lds_offset(&Bs[0][0]), N - n0, K);
  }
  wait_tensorcnt0();
  __syncthreads();
  for (int kk = 0; kk < nk; ++kk) {
    if (kk + 1 < nk && issuer) {
      int k1 = (kk + 1) * 32;
      tdm_load_tile(A  + (size_t)m0 * K + k1, lds_offset(&As[buf ^ 1][0]), M - m0, K);
      tdm_load_tile(Bt + (size_t)n0 * K + k1, lds_offset(&Bs[buf ^ 1][0]), N - n0, K);
    }
    compute_step(&As[buf][0], &Bs[buf][0]);        // overlaps with in-flight TDM
    if (kk + 1 < nk) wait_tensorcnt0();
    __syncthreads();
    buf ^= 1;
  }
#else
  for (int k0 = 0; k0 < K; k0 += 32) {
    #pragma unroll
    for (int it = 0; it < 2; ++it) {
      int c = tid + it * 256;                     // 512 x 16B chunks per operand
      int row = c >> 2, seg = c & 3;
      *(uint4*)&As1[row * 40 + seg * 8] =
          *(const uint4*)(A + (size_t)(m0 + row) * K + k0 + seg * 8);
      uint4 bv = make_uint4(0u, 0u, 0u, 0u);
      int gn = n0 + row;
      if (gn < N) bv = *(const uint4*)(Bt + (size_t)gn * K + k0 + seg * 8);
      *(uint4*)&Bs1[row * 40 + seg * 8] = bv;
    }
    if (k0 + 32 < K) {                            // hint next A tile into cache
      __builtin_prefetch(A + (size_t)(m0 + (tid >> 2)) * K + k0 + 32, 0, 1);
    }
    __syncthreads();
    compute_step(As1, Bs1);
    __syncthreads();
  }
#endif

  // epilogue: bias -> (gelu) -> (+resid) -> store fp32/bf16 (NT for streaming logits)
  #pragma unroll
  for (int j = 0; j < 2; ++j) {
    int gn = n0 + wn * 32 + j * 16 + l16;
    if (gn >= N) continue;
    float bj = bias ? bias[gn] : 0.0f;
    #pragma unroll
    for (int i = 0; i < 4; ++i) {
      #pragma unroll
      for (int r = 0; r < 8; ++r) {
        int gm = m0 + wm * 64 + i * 16 + r + half * 8;
        float v = acc[i][j][r] + bj;
        if (flags & GELU_FLAG) v = 0.5f * v * (1.0f + erff(v * 0.70710678118654752f));
        size_t o = (size_t)gm * N + gn;
        if (resid) v += resid[o];
        if (Cf) {
          if (flags & NT_FLAG) __builtin_nontemporal_store(v, &Cf[o]);
          else                 Cf[o] = v;
        }
        if (Cb) Cb[o] = (__bf16)v;
      }
    }
  }
}

// ---------------- fused flash attention (bf16 WMMA, online softmax) ----------------
// grid: (T/64, H, B); 128 threads = 4 waves; wave w handles q rows qb+w*16..+15.
__global__ __launch_bounds__(128) void attention_kernel(const __bf16* __restrict__ qkv,
                                                        __bf16* __restrict__ y) {
  __shared__ __align__(16) __bf16 Kt[32 * 72];      // [key][e], stride 72
  __shared__ __align__(16) __bf16 Vt[64 * 40];      // [e][key], stride 40 (transposed)
  __shared__ __align__(16) __bf16 Pb[4][16 * 40];   // per-wave P re-stripe buffer
  int tid  = threadIdx.x;
  int qb   = blockIdx.x * 64;
  int h    = blockIdx.y;
  int bB   = blockIdx.z;
  int lane = tid & 31, w = tid >> 5;
  int half = lane >> 4, l16 = lane & 15;

  const size_t rs = 3 * Ee;                         // qkv row stride
  const __bf16* base = qkv + (size_t)bB * Tt * rs;

  // Q fragments (A layout), HD=64 -> two K=32 fragments; stays in registers
  int tq = qb + w * 16 + l16;
  const __bf16* qrow = base + (size_t)tq * rs + h * HDm;
  v16bf q0 = frag_a(qrow + half * 16);
  v16bf q1 = frag_a(qrow + 32 + half * 16);

  v8f o[4];
  float mrun[8], lrun[8];
  #pragma unroll
  for (int e = 0; e < 4; ++e)
    #pragma unroll
    for (int r = 0; r < 8; ++r) o[e][r] = 0.0f;
  #pragma unroll
  for (int r = 0; r < 8; ++r) { mrun[r] = -1e30f; lrun[r] = 0.0f; }

  int ntiles = qb / 32 + 2;                         // keys 0 .. qb+63
  for (int jt = 0; jt < ntiles; ++jt) {
    int j32 = jt * 32;
    __syncthreads();                                // protect Kt/Vt reuse
    // stage K tile [32][64] and V tile transposed [64][32]
    #pragma unroll
    for (int it = 0; it < 2; ++it) {
      int c = tid + it * 128;                       // 256 chunks of 8 bf16
      int key = c >> 3, seg = c & 7;
      const __bf16* kr = base + (size_t)(j32 + key) * rs + Ee + h * HDm + seg * 8;
      *(uint4*)&Kt[key * 72 + seg * 8] = *(const uint4*)kr;
      BF16x8 tv;
      tv.u = *(const uint4*)(base + (size_t)(j32 + key) * rs + 2 * Ee + h * HDm + seg * 8);
      #pragma unroll
      for (int ii = 0; ii < 8; ++ii) Vt[(seg * 8 + ii) * 40 + key] = tv.h[ii];
    }
    __syncthreads();

    // scores S[16 x 32] = Q(16x64) @ K^T(64x32): 4 wmma
    v8f s0, s1;
    #pragma unroll
    for (int r = 0; r < 8; ++r) { s0[r] = 0.0f; s1[r] = 0.0f; }
    v16bf k00 = frag_b(&Kt[l16 * 72 + half * 16]);            // keys 0..15,  e 0..31
    v16bf k01 = frag_b(&Kt[l16 * 72 + 32 + half * 16]);       // keys 0..15,  e 32..63
    v16bf k10 = frag_b(&Kt[(16 + l16) * 72 + half * 16]);     // keys 16..31, e 0..31
    v16bf k11 = frag_b(&Kt[(16 + l16) * 72 + 32 + half * 16]);
    s0 = wmma_bf16(q0, k00, s0); s0 = wmma_bf16(q1, k01, s0);
    s1 = wmma_bf16(q0, k10, s1); s1 = wmma_bf16(q1, k11, s1);

    // online softmax over these 32 keys (per-row stats via 16-lane shuffles)
    #pragma unroll
    for (int r = 0; r < 8; ++r) {
      int qi = qb + w * 16 + r + half * 8;
      float v0s = s0[r] * 0.125f;                   // 1/sqrt(64)
      float v1s = s1[r] * 0.125f;
      if (j32 + l16 > qi)       v0s = -1e30f;       // causal mask
      if (j32 + 16 + l16 > qi)  v1s = -1e30f;
      float mx = fmaxf(v0s, v1s);
      #pragma unroll
      for (int off = 1; off < 16; off <<= 1) mx = fmaxf(mx, __shfl_xor(mx, off, 32));
      float nm = fmaxf(mrun[r], mx);
      float al = __expf(mrun[r] - nm);
      float p0 = __expf(v0s - nm);
      float p1 = __expf(v1s - nm);
      float ps = p0 + p1;
      #pragma unroll
      for (int off = 1; off < 16; off <<= 1) ps += __shfl_xor(ps, off, 32);
      lrun[r] = lrun[r] * al + ps;
      mrun[r] = nm;
      #pragma unroll
      for (int e = 0; e < 4; ++e) o[e][r] *= al;    // rescale running O
      Pb[w][(r + half * 8) * 40 + l16]      = (__bf16)p0;
      Pb[w][(r + half * 8) * 40 + 16 + l16] = (__bf16)p1;
    }
    // re-stripe P into A-fragment layout (wave-local LDS, DS pipe is in-order)
    v16bf pf = frag_a(&Pb[w][l16 * 40 + half * 16]);
    // O(16x64) += P(16x32) @ V(32x64): 4 wmma
    #pragma unroll
    for (int e = 0; e < 4; ++e) {
      v16bf vf = frag_b(&Vt[(e * 16 + l16) * 40 + half * 16]);
      o[e] = wmma_bf16(pf, vf, o[e]);
    }
  }

  // normalize and write y (bf16, [B*T, E])
  #pragma unroll
  for (int e = 0; e < 4; ++e) {
    #pragma unroll
    for (int r = 0; r < 8; ++r) {
      int tqr = qb + w * 16 + r + half * 8;
      float ov = o[e][r] / lrun[r];
      y[((size_t)(bB * Tt + tqr)) * Ee + h * HDm + e * 16 + l16] = (__bf16)ov;
    }
  }
}

// ---------------- loss: per-row log-softmax NLL, then mean ----------------
__global__ void loss_row_kernel(const float* __restrict__ logits, const int* __restrict__ targets,
                                float* __restrict__ nll) {
  int row = blockIdx.x, tid = threadIdx.x;
  __shared__ float red[256];
  const float* lr = logits + (size_t)row * V_;
  float mx = -3.0e38f;
  for (int i = tid; i < V_; i += 256) mx = fmaxf(mx, lr[i]);
  red[tid] = mx; __syncthreads();
  for (int o = 128; o > 0; o >>= 1) { if (tid < o) red[tid] = fmaxf(red[tid], red[tid + o]); __syncthreads(); }
  float M = red[0]; __syncthreads();
  float s = 0.0f;
  for (int i = tid; i < V_; i += 256) s += __expf(lr[i] - M);
  red[tid] = s; __syncthreads();
  for (int o = 128; o > 0; o >>= 1) { if (tid < o) red[tid] += red[tid + o]; __syncthreads(); }
  if (tid == 0) nll[row] = logf(red[0]) + M - lr[targets[row]];
}

__global__ void loss_reduce_kernel(const float* __restrict__ nll, float* __restrict__ out) {
  __shared__ float red[256];
  int tid = threadIdx.x;
  float s = 0.0f;
  for (int i = tid; i < M_ROWS; i += 256) s += nll[i];
  red[tid] = s; __syncthreads();
  for (int o = 128; o > 0; o >>= 1) { if (tid < o) red[tid] += red[tid + o]; __syncthreads(); }
  if (tid == 0) out[0] = red[0] / (float)M_ROWS;
}

// ---------------- host orchestration ----------------
extern "C" void kernel_launch(void* const* d_in, const int* in_sizes, int n_in,
                              void* d_out, int out_size, void* d_ws, size_t ws_size,
                              hipStream_t stream) {
  (void)in_sizes; (void)n_in; (void)out_size; (void)ws_size;
  const int*   idx     = (const int*)d_in[0];
  const int*   targets = (const int*)d_in[1];
  const float* wte     = (const float*)d_in[2];
  const float* wpe     = (const float*)d_in[3];
  const float* ln1_g   = (const float*)d_in[4];
  const float* ln1_b   = (const float*)d_in[5];
  const float* attn_w  = (const float*)d_in[6];
  const float* attn_b  = (const float*)d_in[7];
  const float* proj_w  = (const float*)d_in[8];
  const float* proj_b  = (const float*)d_in[9];
  const float* ln2_g   = (const float*)d_in[10];
  const float* ln2_b   = (const float*)d_in[11];
  const float* fc1_w   = (const float*)d_in[12];
  const float* fc1_b   = (const float*)d_in[13];
  const float* fc2_w   = (const float*)d_in[14];
  const float* fc2_b   = (const float*)d_in[15];
  const float* lnf_g   = (const float*)d_in[16];
  const float* lnf_b   = (const float*)d_in[17];
  float* out = (float*)d_out;

  char* wsp = (char*)d_ws;
  auto carve = [&](size_t bytes) -> void* {
    void* r = (void*)wsp;
    wsp += (bytes + 255) & ~(size_t)255;
    return r;
  };
  float*  x      = (float*) carve((size_t)M_ROWS * Ee * 4);        // residual stream
  __bf16* hbuf   = (__bf16*)carve((size_t)M_ROWS * Ee * 2);        // LN outputs
  __bf16* ybuf   = (__bf16*)carve((size_t)M_ROWS * Ee * 2);        // attention out
  __bf16* qkvb   = (__bf16*)carve((size_t)M_ROWS * 3 * Ee * 2);
  __bf16* ffb    = (__bf16*)carve((size_t)M_ROWS * FF_ * 2);
  float*  nll    = (float*) carve((size_t)M_ROWS * 4);
  __bf16* wteb   = (__bf16*)carve((size_t)V_ * Ee * 2);            // [V][E] (K-contig)
  __bf16* attnWt = (__bf16*)carve((size_t)Ll * 3 * Ee * Ee * 2);   // [3E][E]
  __bf16* projWt = (__bf16*)carve((size_t)Ll * Ee * Ee * 2);       // [E][E]
  __bf16* fc1Wt  = (__bf16*)carve((size_t)Ll * FF_ * Ee * 2);      // [FF][E]
  __bf16* fc2Wt  = (__bf16*)carve((size_t)Ll * Ee * FF_ * 2);      // [E][FF]

  // weight conversion (wte needs no transpose: logits use B^T = wte as-is)
  { int n = V_ * Ee; cvt_bf16_kernel<<<(n + 255) / 256, 256, 0, stream>>>(wte, wteb, n); }
  for (int l = 0; l < Ll; ++l) {
    int n;
    n = Ee * 3 * Ee; transpose_bf16_kernel<<<(n + 255) / 256, 256, 0, stream>>>(
        attn_w + (size_t)l * n, attnWt + (size_t)l * n, Ee, 3 * Ee);
    n = Ee * Ee;     transpose_bf16_kernel<<<(n + 255) / 256, 256, 0, stream>>>(
        proj_w + (size_t)l * n, projWt + (size_t)l * n, Ee, Ee);
    n = Ee * FF_;    transpose_bf16_kernel<<<(n + 255) / 256, 256, 0, stream>>>(
        fc1_w + (size_t)l * n, fc1Wt + (size_t)l * n, Ee, FF_);
    n = FF_ * Ee;    transpose_bf16_kernel<<<(n + 255) / 256, 256, 0, stream>>>(
        fc2_w + (size_t)l * n, fc2Wt + (size_t)l * n, FF_, Ee);
  }

  embed_kernel<<<M_ROWS, 256, 0, stream>>>(idx, wte, wpe, x);

  dim3 blk(256);
  for (int l = 0; l < Ll; ++l) {
    layernorm_bf16_kernel<<<M_ROWS, 256, 0, stream>>>(x, ln1_g + l * Ee, ln1_b + l * Ee, hbuf);
    // qkv = h @ attn_w + attn_b  (bf16 out)
    gemm_bf16_kernel<<<dim3((3 * Ee + 127) / 128, M_ROWS / 128), blk, 0, stream>>>(
        hbuf, attnWt + (size_t)l * 3 * Ee * Ee, attn_b + (size_t)l * 3 * Ee,
        nullptr, nullptr, qkvb, M_ROWS, 3 * Ee, Ee, 0);
    attention_kernel<<<dim3(Tt / 64, Hh, Bb), dim3(128), 0, stream>>>(qkvb, ybuf);
    // x += y @ proj_w + proj_b
    gemm_bf16_kernel<<<dim3((Ee + 127) / 128, M_ROWS / 128), blk, 0, stream>>>(
        ybuf, projWt + (size_t)l * Ee * Ee, proj_b + (size_t)l * Ee,
        x, x, nullptr, M_ROWS, Ee, Ee, 0);
    layernorm_bf16_kernel<<<M_ROWS, 256, 0, stream>>>(x, ln2_g + l * Ee, ln2_b + l * Ee, hbuf);
    // ff = gelu(h @ fc1_w + fc1_b)  (bf16 out)
    gemm_bf16_kernel<<<dim3((FF_ + 127) / 128, M_ROWS / 128), blk, 0, stream>>>(
        hbuf, fc1Wt + (size_t)l * FF_ * Ee, fc1_b + (size_t)l * FF_,
        nullptr, nullptr, ffb, M_ROWS, FF_, Ee, GELU_FLAG);
    // x += ff @ fc2_w + fc2_b
    gemm_bf16_kernel<<<dim3((Ee + 127) / 128, M_ROWS / 128), blk, 0, stream>>>(
        ffb, fc2Wt + (size_t)l * Ee * FF_, fc2_b + (size_t)l * Ee,
        x, x, nullptr, M_ROWS, Ee, FF_, 0);
  }
  layernorm_bf16_kernel<<<M_ROWS, 256, 0, stream>>>(x, lnf_g, lnf_b, hbuf);
  // logits = hf @ wte^T  -> d_out (fp32, non-temporal), ragged N=50257
  gemm_bf16_kernel<<<dim3((V_ + 127) / 128, M_ROWS / 128), blk, 0, stream>>>(
      hbuf, wteb, nullptr, nullptr, out, nullptr, M_ROWS, V_, Ee, NT_FLAG);
  loss_row_kernel<<<M_ROWS, 256, 0, stream>>>(out, targets, nll);
  loss_reduce_kernel<<<1, 256, 0, stream>>>(nll, out + (size_t)M_ROWS * V_);
}
